// FNO_36825049595928
// MI455X (gfx1250) — matrix operand order repaired and checked by reading, not compiled, not run on hardware
//
#include <hip/hip_runtime.h>
#include <math.h>

// ---------------------------------------------------------------------------
// FNO forward for MI455X (gfx1250, wave32, WMMA f16 in / f32 accum).
// Truncated-DFT formulation: all spectral work is dense GEMM on the matrix
// pipes. Internal geometry padded to 272 rows x 288 row-stride so every
// WMMA fragment load/store is an unconditional contiguous b128 access.
// Each GEMM wave owns a full row of N-tiles (4-6 accumulators) so the A
// fragment is loaded once per K-step and amortized over 4-6 WMMAs.
//
//   modesW^T = (X * Fw)^T            M=272,N=64,K=288   (B-major store)
//   modesF   = Fh * modesW           complex: 2 WMMAs w/ sign-swizzled B
//   modesM^T = mode-mix              VALU (M=B=8 too thin; weights coalesced)
//   modesH   = Gh * modesM           complex: 2 WMMAs
//   Y        = modesH * Gw           M=272,N=288,K=64
//   Xnext    = gelu(Y + conv_w*X+b)  pixel-GEMM, pad kept zero via cndmask
// Decoder: 64->128 WMMA + silu -> LDS -> 128->3 VALU dot.
// Workspace ~573 MB (compile-only environment).
// ---------------------------------------------------------------------------

constexpr int NB    = 8;
constexpr int NCIN  = 3;
constexpr int NH    = 256;
constexpr int NW    = 256;
constexpr int NL    = 4;
constexpr int NCW   = 64;
constexpr int NM    = 32;            // retained modes per axis
constexpr int NDOM  = 264;           // H + PAD (FNO domain)
constexpr int HPP   = 272;           // padded rows   (17 * 16)
constexpr int WPS   = 288;           // padded stride (9 * 32)
constexpr int PLANE = HPP * WPS;     // 78336
constexpr int NDEC  = 128;
constexpr int NCOUT = 3;

typedef __attribute__((ext_vector_type(16))) _Float16 v16h;
typedef __attribute__((ext_vector_type(8)))  float    v8f;

__device__ __forceinline__ v8f wmma16(v16h a, v16h b, v8f c) {
  return __builtin_amdgcn_wmma_f32_16x16x32_f16(
      false, a, false, b, (short)0, c, false, false);
}

__device__ __forceinline__ v16h pack16(float4 a, float4 b, float4 c, float4 d) {
  v16h r;
  r[0] = (_Float16)a.x;  r[1] = (_Float16)a.y;  r[2] = (_Float16)a.z;  r[3] = (_Float16)a.w;
  r[4] = (_Float16)b.x;  r[5] = (_Float16)b.y;  r[6] = (_Float16)b.z;  r[7] = (_Float16)b.w;
  r[8] = (_Float16)c.x;  r[9] = (_Float16)c.y;  r[10] = (_Float16)c.z; r[11] = (_Float16)c.w;
  r[12] = (_Float16)d.x; r[13] = (_Float16)d.y; r[14] = (_Float16)d.z; r[15] = (_Float16)d.w;
  return r;
}
__device__ __forceinline__ v16h pack16s(float4 a, float4 b, float4 c, float4 d,
                                        float s) {
  v16h r;
  r[0] = (_Float16)(s * a.x);  r[1] = (_Float16)(s * a.y);
  r[2] = (_Float16)(s * a.z);  r[3] = (_Float16)(s * a.w);
  r[4] = (_Float16)(s * b.x);  r[5] = (_Float16)(s * b.y);
  r[6] = (_Float16)(s * b.z);  r[7] = (_Float16)(s * b.w);
  r[8] = (_Float16)(s * c.x);  r[9] = (_Float16)(s * c.y);
  r[10] = (_Float16)(s * c.z); r[11] = (_Float16)(s * c.w);
  r[12] = (_Float16)(s * d.x); r[13] = (_Float16)(s * d.y);
  r[14] = (_Float16)(s * d.z); r[15] = (_Float16)(s * d.w);
  return r;
}

// A 16x32 fragment, row-major A (ld multiple of 4). Lane holds row m=lane%16;
// halves j<8 -> K=kb+j, j>=8 -> K=kb+16+(j-8), kb = k0 + 8*(lane>=16).
__device__ __forceinline__ v16h load_a(const float* A, int lda, int row0, int k0) {
  int lane = threadIdx.x & 31;
  const float* p = A + (size_t)(row0 + (lane & 15)) * lda + k0 + ((lane >> 4) << 3);
  float4 x0 = *(const float4*)(p);
  float4 x1 = *(const float4*)(p + 4);
  float4 y0 = *(const float4*)(p + 16);
  float4 y1 = *(const float4*)(p + 20);
  return pack16(x0, x1, y0, y1);
}

// A fragment from planar (channel-major) activations: (m=pixel, k=channel).
// Coalesced across lanes (consecutive pixels within a channel plane).
__device__ __forceinline__ v16h load_a_pix(const float* Xb, int pix0, int k0,
                                           size_t plane) {
  int lane = threadIdx.x & 31;
  int kb = k0 + ((lane >> 4) << 3);
  const float* p = Xb + pix0 + (lane & 15);
  v16h a;
#pragma unroll
  for (int j = 0; j < 16; ++j) {
    int k = kb + j + ((j >= 8) ? 8 : 0);
    a[j] = (_Float16)p[(size_t)k * plane];
  }
  return a;
}

// B 32x16 fragment from K-major (transposed) storage Bt[n][k], ld = K stride.
// Lane holds col n=lane%16; j<8 -> K=kb+j, j>=8 -> K=kb+16+(j-8), kb = k0 + 8*(lane>=16).
__device__ __forceinline__ v16h load_bT(const float* Bt, int ldk, int k0, int n0) {
  int lane = threadIdx.x & 31;
  const float* p = Bt + (size_t)(n0 + (lane & 15)) * ldk + k0 + ((lane >> 4) << 3);
  float4 x0 = *(const float4*)(p);
  float4 x1 = *(const float4*)(p + 4);
  float4 x2 = *(const float4*)(p + 16);
  float4 x3 = *(const float4*)(p + 20);
  return pack16(x0, x1, x2, x3);
}

// Swizzled variant for complex multiply with interleaved [re,im] columns:
// even col n -> -Bt[n+1][k], odd col n -> +Bt[n-1][k].
__device__ __forceinline__ v16h load_bT_swz(const float* Bt, int ldk, int k0,
                                            int n0) {
  int lane = threadIdx.x & 31;
  int n = n0 + (lane & 15);
  float sgn = (n & 1) ? 1.0f : -1.0f;
  const float* p = Bt + (size_t)(n ^ 1) * ldk + k0 + ((lane >> 4) << 3);
  float4 x0 = *(const float4*)(p);
  float4 x1 = *(const float4*)(p + 4);
  float4 x2 = *(const float4*)(p + 16);
  float4 x3 = *(const float4*)(p + 20);
  return pack16s(x0, x1, x2, x3, sgn);
}

// C 16x16 f32 tile: VGPR r holds (m = row0 + r + 8*(lane>=16), n = lane%16).
__device__ __forceinline__ void store_c(float* C, int ldc, int row0, int n0,
                                        v8f c) {
  int lane = threadIdx.x & 31;
  float* p = C + (size_t)(row0 + ((lane >> 4) << 3)) * ldc + n0 + (lane & 15);
#pragma unroll
  for (int r = 0; r < 8; ++r) p[(size_t)r * ldc] = c[r];
}
// Transposed C store (writes C^T, contiguous 8 floats per lane).
__device__ __forceinline__ void store_cT(float* Ct, int ldm, int row0, int n0,
                                         v8f c) {
  int lane = threadIdx.x & 31;
  float* p = Ct + (size_t)(n0 + (lane & 15)) * ldm + row0 + ((lane >> 4) << 3);
  *(float4*)(p) = make_float4(c[0], c[1], c[2], c[3]);
  *(float4*)(p + 4) = make_float4(c[4], c[5], c[6], c[7]);
}

__device__ __forceinline__ float gelu_f(float v) {
  return 0.5f * v * (1.0f + erff(v * 0.7071067811865475f));
}
__device__ __forceinline__ float silu_f(float v) {
  return v / (1.0f + __expf(-v));
}

// ---------------------------------------------------------------------------
// Init: DFT matrices (K-major where used as B) + zero modesW pad rows.
// FwT [64][WPS]  : row 2k = cos(2pi k w/264), row 2k+1 = -sin  (fwd along w)
// FhR/FhI [64][WPS]: row r -> kx=(r<32?r:200+r), e^{-i th}      (fwd along h)
// GhR/GhI [HPP][64]: e^{+i th}/264                              (inv along h)
// GwT [WPS][64]  : col 2k = a_k cos/264, col 2k+1 = -a_k sin, a_k=(k?2:1)
// ---------------------------------------------------------------------------
__global__ void k_init(float* FwT, float* FhR, float* FhI, float* GhR,
                       float* GhI, float* GwT, float* MWT) {
  const float twopi = 6.28318530717958647692f;
  int tid = blockIdx.x * blockDim.x + threadIdx.x;
  int stride = gridDim.x * blockDim.x;
  for (int idx = tid; idx < NM * WPS; idx += stride) {
    int k = idx / WPS, w = idx % WPS;
    float th = twopi * (float)((k * w) % NDOM) / (float)NDOM;
    FwT[(2 * k) * WPS + w] = __cosf(th);
    FwT[(2 * k + 1) * WPS + w] = -__sinf(th);
  }
  for (int idx = tid; idx < 64 * WPS; idx += stride) {
    int r = idx / WPS, h = idx % WPS;
    int kx = (r < 32) ? r : (200 + r);
    float th = twopi * (float)((kx * h) % NDOM) / (float)NDOM;
    FhR[r * WPS + h] = __cosf(th);
    FhI[r * WPS + h] = -__sinf(th);
  }
  for (int idx = tid; idx < HPP * 64; idx += stride) {
    int h = idx / 64, r = idx % 64;
    int kx = (r < 32) ? r : (200 + r);
    float th = twopi * (float)((kx * h) % NDOM) / (float)NDOM;
    GhR[idx] = __cosf(th) / (float)NDOM;
    GhI[idx] = __sinf(th) / (float)NDOM;
  }
  for (int idx = tid; idx < WPS * NM; idx += stride) {
    int w = idx / NM, k = idx % NM;
    float a = ((k == 0) ? 1.0f : 2.0f) / (float)NDOM;
    float th = twopi * (float)((k * w) % NDOM) / (float)NDOM;
    GwT[w * 64 + 2 * k] = a * __cosf(th);
    GwT[w * 64 + 2 * k + 1] = -a * __sinf(th);
  }
  // Zero modesW^T pad rows m=272..287 (never written by k_dftW, read by k_dftH).
  for (int idx = tid; idx < NB * NCW * 64 * 16; idx += stride) {
    int bcn = idx >> 4;
    int m = 272 + (idx & 15);
    MWT[(size_t)bcn * WPS + m] = 0.0f;
  }
}

// ---------------------------------------------------------------------------
// Lift: concat coords, 5->32 (gelu) ->64, planar zero-padded X0 [CW][HPP][WPS].
// ---------------------------------------------------------------------------
__global__ void k_lift(const float* __restrict__ x, const float* __restrict__ w1,
                       const float* __restrict__ b1, const float* __restrict__ w2,
                       const float* __restrict__ b2, float* __restrict__ X0) {
  size_t tid = (size_t)blockIdx.x * blockDim.x + threadIdx.x;
  int wp = (int)(tid % WPS);
  int hp = (int)((tid / WPS) % HPP);
  int b = (int)(tid / PLANE);
  size_t obase = (size_t)b * NCW * PLANE + (size_t)hp * WPS + wp;
  if (hp < NH && wp < NW) {
    float in[5];
#pragma unroll
    for (int c = 0; c < NCIN; ++c)
      in[c] = x[(((size_t)b * NCIN + c) * NH + hp) * NW + wp];
    in[3] = (float)hp / (float)(NH - 1);
    in[4] = (float)wp / (float)(NW - 1);
    float hid[32];
#pragma unroll
    for (int o = 0; o < 32; ++o) {
      float s = b1[o];
#pragma unroll
      for (int i = 0; i < 5; ++i) s += w1[o * 5 + i] * in[i];
      hid[o] = gelu_f(s);
    }
    for (int c = 0; c < NCW; ++c) {
      float s = b2[c];
#pragma unroll
      for (int o = 0; o < 32; ++o) s += w2[c * 32 + o] * hid[o];
      X0[obase + (size_t)c * PLANE] = s;
    }
  } else {
    for (int c = 0; c < NCW; ++c) X0[obase + (size_t)c * PLANE] = 0.0f;
  }
}

// ---------------------------------------------------------------------------
// modesW^T[bc][n][h] = sum_w X[bc][h][w] * FwT[n][w]   M=272,N=64,K=288.
// One wave owns one M-tile and all 4 N-tiles: A loaded once per K-step.
// ---------------------------------------------------------------------------
__global__ void k_dftW(const float* __restrict__ X, const float* __restrict__ FwT,
                       float* __restrict__ MWT) {
  int wave = blockIdx.x * (blockDim.x >> 5) + (threadIdx.x >> 5);
  int bc = wave / 17;
  if (bc >= NB * NCW) return;
  int mt = wave % 17;
  const float* A = X + (size_t)bc * PLANE;
  float* Ct = MWT + (size_t)bc * 64 * WPS;
  v8f acc[4] = {};
  for (int k0 = 0; k0 < WPS; k0 += 32) {
    v16h a = load_a(A, WPS, mt * 16, k0);
#pragma unroll
    for (int nt = 0; nt < 4; ++nt) {
      v16h b = load_bT(FwT, WPS, k0, nt * 16);
      acc[nt] = wmma16(a, b, acc[nt]);
    }
  }
#pragma unroll
  for (int nt = 0; nt < 4; ++nt) store_cT(Ct, WPS, mt * 16, nt * 16, acc[nt]);
}

// ---------------------------------------------------------------------------
// modesF[bc][kx][n] = sum_h Fh[kx][h]*modesW[h][n]  complex via 2 WMMAs.
// M=64,N=64,K=288.  B = modesW^T (K-major, ld=WPS). 4 N-tiles per wave.
// ---------------------------------------------------------------------------
__global__ void k_dftH(const float* __restrict__ MWT, const float* __restrict__ FhR,
                       const float* __restrict__ FhI, float* __restrict__ MF) {
  int wave = blockIdx.x * (blockDim.x >> 5) + (threadIdx.x >> 5);
  int bc = wave / 4;
  if (bc >= NB * NCW) return;
  int mt = wave % 4;
  const float* Bt = MWT + (size_t)bc * 64 * WPS;
  float* C = MF + (size_t)bc * 64 * 64;
  v8f acc[4] = {};
  for (int k0 = 0; k0 < WPS; k0 += 32) {
    v16h ar = load_a(FhR, WPS, mt * 16, k0);
    v16h ai = load_a(FhI, WPS, mt * 16, k0);
#pragma unroll
    for (int nt = 0; nt < 4; ++nt) {
      v16h bw = load_bT(Bt, WPS, k0, nt * 16);
      acc[nt] = wmma16(ar, bw, acc[nt]);
      v16h bs = load_bT_swz(Bt, WPS, k0, nt * 16);
      acc[nt] = wmma16(ai, bs, acc[nt]);
    }
  }
#pragma unroll
  for (int nt = 0; nt < 4; ++nt) store_c(C, 64, mt * 16, nt * 16, acc[nt]);
}

// ---------------------------------------------------------------------------
// Mode mixing (per-mode complex 64x64, batch B=8). VALU; lanes map to
// contiguous ky so weight (float2) and data (float2) loads coalesce.
// Output stored K-major for k_idftH: MMT[bc][n=2ky+ri][kx].
// ---------------------------------------------------------------------------
__global__ void k_mix(const float* __restrict__ MF, const float* __restrict__ w1,
                      const float* __restrict__ w2, float* __restrict__ MMT) {
  int tid = blockIdx.x * blockDim.x + threadIdx.x;
  if (tid >= NCW * 64 * NM) return;
  int ky = tid & 31;
  int kx = (tid >> 5) & 63;
  int o = tid >> 11;
  const float* wsel = (kx < 32) ? w1 : w2;
  int mx = (kx < 32) ? kx : (kx - 32);
  float accR[NB] = {}, accI[NB] = {};
  for (int i = 0; i < NCW; ++i) {
    float2 wv = *(const float2*)(wsel +
        ((((size_t)i * NCW + o) * NM + mx) * NM + ky) * 2);
    const float* ip = MF + (((size_t)i * 64 + kx) * 64) + 2 * ky;
#pragma unroll
    for (int b = 0; b < NB; ++b) {
      float2 xv = *(const float2*)(ip + (size_t)b * NCW * 4096);
      accR[b] += xv.x * wv.x - xv.y * wv.y;
      accI[b] += xv.x * wv.y + xv.y * wv.x;
    }
  }
#pragma unroll
  for (int b = 0; b < NB; ++b) {
    float* op = MMT + (size_t)(b * NCW + o) * 4096;
    op[(2 * ky) * 64 + kx] = accR[b];
    op[(2 * ky + 1) * 64 + kx] = accI[b];
  }
}

// ---------------------------------------------------------------------------
// modesH[bc][h][n] = sum_kx Gh[h][kx]*MM[kx][n]  complex. M=272,N=64,K=64.
// 4 N-tiles per wave.
// ---------------------------------------------------------------------------
__global__ void k_idftH(const float* __restrict__ MMT, const float* __restrict__ GhR,
                        const float* __restrict__ GhI, float* __restrict__ MH) {
  int wave = blockIdx.x * (blockDim.x >> 5) + (threadIdx.x >> 5);
  int bc = wave / 17;
  if (bc >= NB * NCW) return;
  int mt = wave % 17;
  const float* Bt = MMT + (size_t)bc * 4096;
  float* C = MH + (size_t)bc * HPP * 64;
  v8f acc[4] = {};
  for (int k0 = 0; k0 < 64; k0 += 32) {
    v16h ar = load_a(GhR, 64, mt * 16, k0);
    v16h ai = load_a(GhI, 64, mt * 16, k0);
#pragma unroll
    for (int nt = 0; nt < 4; ++nt) {
      v16h bw = load_bT(Bt, 64, k0, nt * 16);
      acc[nt] = wmma16(ar, bw, acc[nt]);
      v16h bs = load_bT_swz(Bt, 64, k0, nt * 16);
      acc[nt] = wmma16(ai, bs, acc[nt]);
    }
  }
#pragma unroll
  for (int nt = 0; nt < 4; ++nt) store_c(C, 64, mt * 16, nt * 16, acc[nt]);
}

// ---------------------------------------------------------------------------
// Y[bc][h][w] = sum_kk modesH[h][kk]*Gw[kk][w]   M=272,N=288,K=64.
// B = GwT (K-major, ld=64). 6 N-tiles per wave (3 groups of 6 cover N=288).
// ---------------------------------------------------------------------------
__global__ void k_idftW(const float* __restrict__ MH, const float* __restrict__ GwT,
                        float* __restrict__ Y) {
  int wave = blockIdx.x * (blockDim.x >> 5) + (threadIdx.x >> 5);
  int bc = wave / 51;  // 17 m-tiles * 3 n-groups
  if (bc >= NB * NCW) return;
  int t = wave % 51;
  int mt = t / 3, ng = t % 3;
  int n0 = ng * 96;
  const float* A = MH + (size_t)bc * HPP * 64;
  float* C = Y + (size_t)bc * PLANE;
  v8f acc[6] = {};
  for (int k0 = 0; k0 < 64; k0 += 32) {
    v16h a = load_a(A, 64, mt * 16, k0);
#pragma unroll
    for (int nt = 0; nt < 6; ++nt) {
      v16h b = load_bT(GwT, 64, k0, n0 + nt * 16);
      acc[nt] = wmma16(a, b, acc[nt]);
    }
  }
#pragma unroll
  for (int nt = 0; nt < 6; ++nt)
    store_c(C, WPS, mt * 16, n0 + nt * 16, acc[nt]);
}

// ---------------------------------------------------------------------------
// Xn = act(Y + conv_w*Xc + b) on the 264x264 domain; 0 in the extra pad
// (branchless select). M=16 pixels, K=64, all 4 N-tiles per wave.
// ---------------------------------------------------------------------------
__global__ void k_pw(const float* __restrict__ Xc, const float* __restrict__ Y,
                     const float* __restrict__ cw, const float* __restrict__ cb,
                     float* __restrict__ Xn, int apply_act) {
  const int MT = PLANE / 16;  // 4896
  int wave = blockIdx.x * (blockDim.x >> 5) + (threadIdx.x >> 5);
  int b = wave / MT;
  if (b >= NB) return;
  int mt = wave % MT;
  int pix0 = mt * 16;
  const float* Xb = Xc + (size_t)b * NCW * PLANE;
  v8f acc[4] = {};
  for (int k0 = 0; k0 < NCW; k0 += 32) {
    v16h a = load_a_pix(Xb, pix0, k0, (size_t)PLANE);
#pragma unroll
    for (int nt = 0; nt < 4; ++nt) {
      // conv_w slice is (o,i)-major: B[k=i][n=o] = cw[o*64+i] -> K-major.
      v16h w = load_bT(cw, NCW, k0, nt * 16);
      acc[nt] = wmma16(a, w, acc[nt]);
    }
  }
  int lane = threadIdx.x & 31;
  int hh = pix0 / WPS;              // 288 % 16 == 0: tile stays in one row
  int w0 = (pix0 % WPS) + ((lane >> 4) << 3);
  int mb = pix0 + ((lane >> 4) << 3);
  bool hok = (hh < NDOM);
#pragma unroll
  for (int nt = 0; nt < 4; ++nt) {
    int n = nt * 16 + (lane & 15);
    float bias = cb[n];
    const float* yp = Y + ((size_t)(b * NCW + n)) * PLANE + mb;
    float* op = Xn + ((size_t)(b * NCW + n)) * PLANE + mb;
#pragma unroll
    for (int r = 0; r < 8; ++r) {
      float v = acc[nt][r] + bias + yp[r];
      v = apply_act ? gelu_f(v) : v;
      bool ok = hok && (w0 + r < NDOM);
      op[r] = ok ? v : 0.0f;
    }
  }
}

// ---------------------------------------------------------------------------
// Decoder: 16 cropped pixels/wave. WMMA 64->128, silu -> LDS, 128->3 VALU.
// blockDim=128 (4 waves), LDS 32 KB, exact grid (no early exit).
// ---------------------------------------------------------------------------
__global__ void k_decoder(const float* __restrict__ X, const float* __restrict__ w1,
                          const float* __restrict__ b1, const float* __restrict__ w2,
                          const float* __restrict__ b2, float* __restrict__ out) {
  __shared__ float lds[4][16 * NDEC];
  int widx = threadIdx.x >> 5;
  int lane = threadIdx.x & 31;
  int wave = blockIdx.x * 4 + widx;
  const int tilesPerB = (NH * NW) / 16;  // 4096
  int b = wave / tilesPerB;
  int pt = wave % tilesPerB;
  int pix0 = pt * 16;

  v8f acc[8] = {};
  const float* Xb = X + (size_t)b * NCW * PLANE;
  for (int k0 = 0; k0 < NCW; k0 += 32) {
    int m = lane & 15;
    int p = pix0 + m;
    int hh = p / NW, ww = p % NW;
    const float* xp = Xb + (size_t)hh * WPS + ww;
    int kb = k0 + ((lane >> 4) << 3);
    v16h a;
#pragma unroll
    for (int j = 0; j < 16; ++j) {
      int k = kb + j + ((j >= 8) ? 8 : 0);
      a[j] = (_Float16)xp[(size_t)k * PLANE];
    }
#pragma unroll
    for (int ntile = 0; ntile < 8; ++ntile) {
      v16h w = load_bT(w1, NCW, k0, ntile * 16);  // dec_w1 (DEC,CW) K-major
      acc[ntile] = wmma16(a, w, acc[ntile]);
    }
  }
  float* L = &lds[widx][0];
#pragma unroll
  for (int ntile = 0; ntile < 8; ++ntile) {
    int n = ntile * 16 + (lane & 15);
    float bb = b1[n];
    int mb = (lane >> 4) << 3;
#pragma unroll
    for (int r = 0; r < 8; ++r) {
      L[(mb + r) * NDEC + n] = silu_f(acc[ntile][r] + bb);
    }
  }
  __syncthreads();
  if (lane < 16) {
    int p = pix0 + lane;
    int hh = p / NW, ww = p % NW;
    const float* hp = &L[lane * NDEC];
#pragma unroll
    for (int o = 0; o < NCOUT; ++o) {
      float s = b2[o];
      for (int d = 0; d < NDEC; ++d) s += w2[o * NDEC + d] * hp[d];
      out[(((size_t)b * NCOUT + o) * NH + hh) * NW + ww] = s;
    }
  }
}

// ---------------------------------------------------------------------------
extern "C" void kernel_launch(void* const* d_in, const int* in_sizes, int n_in,
                              void* d_out, int out_size, void* d_ws,
                              size_t ws_size, hipStream_t stream) {
  (void)in_sizes; (void)n_in; (void)out_size; (void)ws_size;
  const float* x    = (const float*)d_in[0];
  const float* lw1  = (const float*)d_in[1];
  const float* lb1  = (const float*)d_in[2];
  const float* lw2  = (const float*)d_in[3];
  const float* lb2  = (const float*)d_in[4];
  const float* spw1 = (const float*)d_in[5];
  const float* spw2 = (const float*)d_in[6];
  const float* cw   = (const float*)d_in[7];
  const float* cb   = (const float*)d_in[8];
  const float* dw1  = (const float*)d_in[9];
  const float* db1  = (const float*)d_in[10];
  const float* dw2  = (const float*)d_in[11];
  const float* db2  = (const float*)d_in[12];
  float* out = (float*)d_out;

  const size_t SZX = (size_t)NB * NCW * PLANE;       // 40,108,032
  const size_t SZW = (size_t)NB * NCW * 64 * WPS;    //  9,437,184
  const size_t SZF = (size_t)NB * NCW * 64 * 64;     //  2,097,152
  const size_t SZH = (size_t)NB * NCW * HPP * 64;    //  8,912,896
  float* p = (float*)d_ws;
  float* Xa  = p; p += SZX;
  float* Xb  = p; p += SZX;
  float* Ys  = p; p += SZX;
  float* MWT = p; p += SZW;
  float* MF  = p; p += SZF;
  float* MMT = p; p += SZF;
  float* MH  = p; p += SZH;
  float* FwT = p; p += (size_t)64 * WPS;
  float* FhR = p; p += (size_t)64 * WPS;
  float* FhI = p; p += (size_t)64 * WPS;
  float* GhR = p; p += (size_t)HPP * 64;
  float* GhI = p; p += (size_t)HPP * 64;
  float* GwT = p; p += (size_t)WPS * 64;

  k_init<<<128, 256, 0, stream>>>(FwT, FhR, FhI, GhR, GhI, GwT, MWT);
  k_lift<<<(NB * PLANE) / 256, 256, 0, stream>>>(x, lw1, lb1, lw2, lb2, Xa);
  float* Xc = Xa;
  float* Xn = Xb;
  for (int l = 0; l < NL; ++l) {
    k_dftW<<<1088, 256, 0, stream>>>(Xc, FwT, MWT);       // 512*17 waves
    k_dftH<<<256, 256, 0, stream>>>(MWT, FhR, FhI, MF);   // 512*4 waves
    k_mix<<<512, 256, 0, stream>>>(MF,
        spw1 + (size_t)l * NCW * NCW * NM * NM * 2,
        spw2 + (size_t)l * NCW * NCW * NM * NM * 2, MMT);
    k_idftH<<<1088, 256, 0, stream>>>(MMT, GhR, GhI, MH); // 512*17 waves
    k_idftW<<<3264, 256, 0, stream>>>(MH, GwT, Ys);       // 512*51 waves
    k_pw<<<4896, 256, 0, stream>>>(Xc, Ys, cw + (size_t)l * NCW * NCW,
                                   cb + (size_t)l * NCW, Xn,
                                   (l < NL - 1) ? 1 : 0);
    float* tswap = Xc; Xc = Xn; Xn = tswap;
  }
  k_decoder<<<8192, 128, 0, stream>>>(Xc, dw1, db1, dw2, db2, out);
}
